// GPT2RSMModel_53266184405093
// MI455X (gfx1250) — matrix operand order repaired and checked by reading, not compiled, not run on hardware
//
#include <hip/hip_runtime.h>

// Model constants (match reference)
#define BB 4
#define TT 256
#define DD 1024
#define LL 4
#define NS 64
#define FF 4096
#define VV 32000
#define EPSF 1e-5f
#define SCALEF 0.03125f   // 1/sqrt(1024)
#define EMAF 0.9f

typedef __attribute__((ext_vector_type(16))) __bf16 v16bf;
typedef __attribute__((ext_vector_type(8)))  float  v8f;

__device__ __forceinline__ float gelu_tanh(float x) {
  float x3 = x * x * x;
  return 0.5f * x * (1.0f + tanhf(0.7978845608028654f * (x + 0.044715f * x3)));
}

// ---------------------------------------------------------------------------
// WMMA GEMM: C[M,N] = act(A[M,K] * W + bias)
//   !TRANSB: W is K x N row-major.   TRANSB: W is N x K row-major (lm_head).
// Block = 128 threads (4 waves). Tile = 32(M) x 64(N), K-step 32.
// Tiles are staged into LDS *in WMMA fragment order* (contiguous k = 16h+e maps applied
// at store time), so each lane's v16bf operand is one aligned 32B LDS load
// (ds_load_b128 pairs) instead of 16 scalar ds_load_u16 picks.
// Each wave owns one 16-wide N strip and both 16-row M strips: 2 WMMAs/K-step
// sharing a single B fragment.
// ---------------------------------------------------------------------------
template <bool TRANSB, bool BIAS, bool GELU>
__global__ __launch_bounds__(128)
void gemm_wmma(const float* __restrict__ A, const float* __restrict__ W,
               const float* __restrict__ bias, float* __restrict__ C,
               int M, int N, int K) {
  // Fragment-order staging buffers: [.. ][lane][element]
  __shared__ __align__(32) __bf16 Af[2][32][16];  // two 16-row M subtiles, 2 KB
  __shared__ __align__(32) __bf16 Bf[4][32][16];  // one per wave,           4 KB

  const int tid  = threadIdx.x;
  const int lane = tid & 31;
  const int wid  = tid >> 5;
  const int half = lane >> 4;
  const int l16  = lane & 15;
  const int n0   = blockIdx.x * 64;
  const int m0   = blockIdx.y * 32;

  v8f acc0 = {};
  v8f acc1 = {};

  for (int kk = 0; kk < K; kk += 32) {
    // --- stage A tile (32 x 32) as float4 loads, scatter in fragment order ---
    // A map: element e of lane (h, mr) holds column ka = h*16 + e
    // inverse: h = c>>4 ; e = c&15 ; lane = h*16 + mr
    for (int i = tid; i < (32 * 32) / 4; i += 128) {
      int idx = i * 4;
      int r = idx >> 5, c0 = idx & 31;
      int gm = m0 + r;
      float4 v = make_float4(0.f, 0.f, 0.f, 0.f);
      if (gm < M) v = *reinterpret_cast<const float4*>(&A[(size_t)gm * K + kk + c0]);
      float tmp4[4] = {v.x, v.y, v.z, v.w};
#pragma unroll
      for (int j = 0; j < 4; ++j) {
        int c = c0 + j;
        int h = c >> 4;
        int e = c & 15;
        Af[r >> 4][h * 16 + (r & 15)][e] = (__bf16)tmp4[j];
      }
    }
    // --- stage B tile (32 x 64) in fragment order ---
    // B map: element e of lane (h, l16) in wave w holds (k = h*16 + e, n = w*16 + l16)
    if (!TRANSB) {
      // W row-major K x N: contiguous along n -> float4 along columns
      for (int i = tid; i < (32 * 64) / 4; i += 128) {
        int idx = i * 4;
        int k = idx >> 6, c0 = idx & 63;
        int gk = kk + k, gn0 = n0 + c0;
        float4 v = make_float4(0.f, 0.f, 0.f, 0.f);
        if (gn0 + 3 < N) {
          v = *reinterpret_cast<const float4*>(&W[(size_t)gk * N + gn0]);
        } else {
          float* p = &v.x;
          for (int j = 0; j < 4; ++j)
            if (gn0 + j < N) p[j] = W[(size_t)gk * N + gn0 + j];
        }
        float tmp4[4] = {v.x, v.y, v.z, v.w};
#pragma unroll
        for (int j = 0; j < 4; ++j) {
          int c = c0 + j;
          Bf[c >> 4][(k >> 4) * 16 + (c & 15)][k & 15] = (__bf16)tmp4[j];
        }
      }
    } else {
      // W row-major N x K: contiguous along k -> float4 along k
      for (int i = tid; i < (64 * 32) / 4; i += 128) {
        int n = i & 63;          // column within tile
        int k0 = (i >> 6) * 4;   // 4 consecutive k
        int gn = n0 + n;
        float4 v = make_float4(0.f, 0.f, 0.f, 0.f);
        if (gn < N) v = *reinterpret_cast<const float4*>(&W[(size_t)gn * K + kk + k0]);
        float tmp4[4] = {v.x, v.y, v.z, v.w};
#pragma unroll
        for (int j = 0; j < 4; ++j) {
          int k = k0 + j;
          Bf[n >> 4][(k >> 4) * 16 + (n & 15)][k & 15] = (__bf16)tmp4[j];
        }
      }
    }
    if (kk + 32 < K)  // pull next A tile toward L2 (global_prefetch_b8)
      __builtin_prefetch(&A[(size_t)m0 * K + kk + 32], 0, 0);
    __syncthreads();

    // --- one aligned 32B LDS load per operand ---
    v16bf a0 = *reinterpret_cast<const v16bf*>(&Af[0][lane][0]);
    v16bf a1 = *reinterpret_cast<const v16bf*>(&Af[1][lane][0]);
    v16bf b  = *reinterpret_cast<const v16bf*>(&Bf[wid][lane][0]);
    acc0 = __builtin_amdgcn_wmma_f32_16x16x32_bf16(false, a0, false, b,
                                                   (short)0, acc0, false, false);
    acc1 = __builtin_amdgcn_wmma_f32_16x16x32_bf16(false, a1, false, b,
                                                   (short)0, acc1, false, false);
    __syncthreads();
  }

  // --- store C: lane = column n, VGPR v = row (v + 8*half) in each subtile ---
  const int gn = n0 + wid * 16 + l16;
  if (gn < N) {
    float bv = BIAS ? bias[gn] : 0.0f;
#pragma unroll
    for (int v = 0; v < 8; ++v) {
      int gm = m0 + v + 8 * half;
      if (gm < M) {
        float r = acc0[v] + bv;
        if (GELU) r = gelu_tanh(r);
        C[(size_t)gm * N + gn] = r;
      }
      int gm1 = gm + 16;
      if (gm1 < M) {
        float r = acc1[v] + bv;
        if (GELU) r = gelu_tanh(r);
        C[(size_t)gm1 * N + gn] = r;
      }
    }
  }
}

// ---------------------------------------------------------------------------
// Elementwise / reduction kernels (256-thread blocks, wave32 shuffles)
// ---------------------------------------------------------------------------
__device__ __forceinline__ float block_sum_256(float v, float* red) {
  for (int off = 16; off > 0; off >>= 1) v += __shfl_xor(v, off, 32);
  int lane = threadIdx.x & 31, wid = threadIdx.x >> 5;
  if (lane == 0) red[wid] = v;
  __syncthreads();
  float r = (threadIdx.x < 8) ? red[threadIdx.x] : 0.0f;
  if (wid == 0) {
    for (int off = 4; off > 0; off >>= 1) r += __shfl_xor(r, off, 32);
    if (lane == 0) red[0] = r;
  }
  __syncthreads();
  float out = red[0];
  __syncthreads();
  return out;
}

// h[b,t,:] = wte[ids[b,t],:] + wpe[t,:]
__global__ __launch_bounds__(256)
void embed_kernel(const int* __restrict__ ids, const float* __restrict__ wte,
                  const float* __restrict__ wpe, float* __restrict__ h) {
  int bt = blockIdx.x;
  int t = bt % TT;
  int id = ids[bt];
  for (int d = threadIdx.x; d < DD; d += 256)
    h[(size_t)bt * DD + d] = wte[(size_t)id * DD + d] + wpe[(size_t)t * DD + d];
}

// states[l,b,n,:] = slot_init[l,n,:]
__global__ __launch_bounds__(256)
void init_states_kernel(const float* __restrict__ slot_init, float* __restrict__ states) {
  size_t total = (size_t)LL * BB * NS * DD;
  for (size_t i = (size_t)blockIdx.x * blockDim.x + threadIdx.x; i < total;
       i += (size_t)gridDim.x * blockDim.x) {
    size_t d = i & (DD - 1);
    size_t n = (i >> 10) & (NS - 1);
    size_t l = i / ((size_t)BB * NS * DD);
    states[i] = slot_init[(l * NS + n) * DD + d];
  }
}

// slot write: softmax addressing + EMA update (one block per batch element)
__global__ __launch_bounds__(256)
void write_update_kernel(float* __restrict__ st, const float* __restrict__ kproj,
                         const float* __restrict__ vproj) {
  __shared__ float wl[NS];
  const int b = blockIdx.x;
  const int lane = threadIdx.x & 31, wid = threadIdx.x >> 5;
  for (int n = wid * 8; n < wid * 8 + 8; ++n) {
    const float* sp = st + ((size_t)b * NS + n) * DD;
    const float* kp = kproj + (size_t)b * DD;
    float s = 0.0f;
    for (int d = lane; d < DD; d += 32) s += sp[d] * kp[d];
    for (int off = 16; off > 0; off >>= 1) s += __shfl_xor(s, off, 32);
    if (lane == 0) wl[n] = s * SCALEF;
  }
  __syncthreads();
  if (threadIdx.x == 0) {
    float mx = -1e30f;
    for (int n = 0; n < NS; ++n) mx = fmaxf(mx, wl[n]);
    float sum = 0.0f;
    for (int n = 0; n < NS; ++n) { wl[n] = __expf(wl[n] - mx); sum += wl[n]; }
    float inv = 1.0f / sum;
    for (int n = 0; n < NS; ++n) wl[n] *= inv;
  }
  __syncthreads();
  for (int e = threadIdx.x; e < NS * DD; e += 256) {
    int n = e >> 10, d = e & (DD - 1);
    size_t idx = ((size_t)b * NS + n) * DD + d;
    st[idx] = EMAF * st[idx] + (1.0f - EMAF) * wl[n] * vproj[(size_t)b * DD + d];
  }
}

// ctx[b,:] = softmax(q[b]·keys[b,n]/sqrt(D)) @ vals[b]
__global__ __launch_bounds__(256)
void attn_ctx_kernel(const float* __restrict__ q, const float* __restrict__ keys,
                     const float* __restrict__ vals, float* __restrict__ ctx) {
  __shared__ float att[NS];
  const int b = blockIdx.x;
  const int lane = threadIdx.x & 31, wid = threadIdx.x >> 5;
  for (int n = wid * 8; n < wid * 8 + 8; ++n) {
    const float* kp = keys + ((size_t)b * NS + n) * DD;
    const float* qp = q + (size_t)b * DD;
    float s = 0.0f;
    for (int d = lane; d < DD; d += 32) s += kp[d] * qp[d];
    for (int off = 16; off > 0; off >>= 1) s += __shfl_xor(s, off, 32);
    if (lane == 0) att[n] = s * SCALEF;
  }
  __syncthreads();
  if (threadIdx.x == 0) {
    float mx = -1e30f;
    for (int n = 0; n < NS; ++n) mx = fmaxf(mx, att[n]);
    float sum = 0.0f;
    for (int n = 0; n < NS; ++n) { att[n] = __expf(att[n] - mx); sum += att[n]; }
    float inv = 1.0f / sum;
    for (int n = 0; n < NS; ++n) att[n] *= inv;
  }
  __syncthreads();
  for (int d = threadIdx.x; d < DD; d += 256) {
    float c = 0.0f;
    for (int n = 0; n < NS; ++n)
      c += att[n] * vals[((size_t)b * NS + n) * DD + d];
    ctx[(size_t)b * DD + d] = c;
  }
}

// y = LN(a + t) * g + beta ; optionally write sum (a+t) to sum_out. t may be null.
__global__ __launch_bounds__(256)
void add_ln_kernel(const float* __restrict__ a, const float* __restrict__ t,
                   float* __restrict__ sum_out, float* __restrict__ y,
                   const float* __restrict__ g, const float* __restrict__ beta) {
  __shared__ float buf[DD];
  __shared__ float red[8];
  const size_t row = blockIdx.x;
  for (int d = threadIdx.x; d < DD; d += 256) {
    float v = a[row * DD + d];
    if (t) v += t[row * DD + d];
    buf[d] = v;
  }
  __syncthreads();
  float s = 0.0f;
  for (int d = threadIdx.x; d < DD; d += 256) s += buf[d];
  float mean = block_sum_256(s, red) * (1.0f / DD);
  float vs = 0.0f;
  for (int d = threadIdx.x; d < DD; d += 256) {
    float dx = buf[d] - mean;
    vs += dx * dx;
  }
  float var = block_sum_256(vs, red) * (1.0f / DD);
  float rstd = rsqrtf(var + EPSF);
  for (int d = threadIdx.x; d < DD; d += 256) {
    float xv = (buf[d] - mean) * rstd;
    y[row * DD + d] = xv * g[d] + beta[d];
    if (sum_out) sum_out[row * DD + d] = buf[d];
  }
}

__global__ __launch_bounds__(256)
void vecadd_kernel(float* __restrict__ x, const float* __restrict__ t, int n) {
  int i = blockIdx.x * blockDim.x + threadIdx.x;
  if (i < n) x[i] += t[i];
}

__global__ __launch_bounds__(256)
void gather_x_kernel(const float* __restrict__ h, float* __restrict__ x, int t) {
  int b = blockIdx.x;
  for (int d = threadIdx.x; d < DD; d += 256)
    x[(size_t)b * DD + d] = h[((size_t)b * TT + t) * DD + d];
}

__global__ __launch_bounds__(256)
void scatter_x_kernel(const float* __restrict__ x, float* __restrict__ outbuf, int t) {
  int b = blockIdx.x;
  for (int d = threadIdx.x; d < DD; d += 256)
    outbuf[((size_t)b * TT + t) * DD + d] = x[(size_t)b * DD + d];
}

// ---------------------------------------------------------------------------
// Host-side dispatch
// ---------------------------------------------------------------------------
static void launch_gemm(bool transb, bool biasf, bool geluf, const float* A,
                        const float* W, const float* bias, float* C, int M, int N,
                        int K, hipStream_t s) {
  dim3 grid((N + 63) / 64, (M + 31) / 32);
  dim3 blk(128);
  if (transb)
    gemm_wmma<true, false, false><<<grid, blk, 0, s>>>(A, W, bias, C, M, N, K);
  else if (biasf && geluf)
    gemm_wmma<false, true, true><<<grid, blk, 0, s>>>(A, W, bias, C, M, N, K);
  else if (biasf)
    gemm_wmma<false, true, false><<<grid, blk, 0, s>>>(A, W, bias, C, M, N, K);
  else
    gemm_wmma<false, false, false><<<grid, blk, 0, s>>>(A, W, bias, C, M, N, K);
}

extern "C" void kernel_launch(void* const* d_in, const int* in_sizes, int n_in,
                              void* d_out, int out_size, void* d_ws, size_t ws_size,
                              hipStream_t stream) {
  (void)in_sizes; (void)n_in; (void)out_size; (void)ws_size;

  const int*   ids       = (const int*)d_in[0];
  const float* wte       = (const float*)d_in[1];
  const float* wpe       = (const float*)d_in[2];
  const float* slot_init = (const float*)d_in[3];
  const float* Wk        = (const float*)d_in[4];
  const float* Wv        = (const float*)d_in[5];
  const float* Wq        = (const float*)d_in[6];
  const float* ln1_g     = (const float*)d_in[7];
  const float* ln1_b     = (const float*)d_in[8];
  const float* ffn1_w    = (const float*)d_in[9];
  const float* ffn1_b    = (const float*)d_in[10];
  const float* ffn2_w    = (const float*)d_in[11];
  const float* ffn2_b    = (const float*)d_in[12];
  const float* ln2t_g    = (const float*)d_in[13];
  const float* ln2t_b    = (const float*)d_in[14];
  const float* Rk        = (const float*)d_in[15];
  const float* Rv        = (const float*)d_in[16];
  const float* out_w     = (const float*)d_in[17];
  const float* out_b     = (const float*)d_in[18];
  const float* lnm_g     = (const float*)d_in[19];
  const float* lnm_b     = (const float*)d_in[20];
  const float* mlp1_w    = (const float*)d_in[21];
  const float* mlp1_b    = (const float*)d_in[22];
  const float* mlp2_w    = (const float*)d_in[23];
  const float* mlp2_b    = (const float*)d_in[24];
  const float* lnf_g     = (const float*)d_in[25];
  const float* lnf_b     = (const float*)d_in[26];
  float* logits = (float*)d_out;
  float* ws = (float*)d_ws;

  // workspace layout (floats)
  size_t off = 0;
  float* h      = ws + off; off += (size_t)BB * TT * DD;   // reused as lnout later
  float* states = ws + off; off += (size_t)LL * BB * NS * DD;
  float* outbuf = ws + off; off += (size_t)BB * TT * DD;
  float* keys   = ws + off; off += (size_t)BB * NS * DD;
  float* vals   = ws + off; off += (size_t)BB * NS * DD;
  float* x      = ws + off; off += (size_t)BB * DD;
  float* kproj  = ws + off; off += (size_t)BB * DD;
  float* vproj  = ws + off; off += (size_t)BB * DD;
  float* q      = ws + off; off += (size_t)BB * DD;
  float* ctx    = ws + off; off += (size_t)BB * DD;
  float* ff1    = ws + off; off += (size_t)BB * FF;
  float* ff2    = ws + off; off += (size_t)BB * DD;
  float* tmp    = ws + off; off += (size_t)BB * DD;
  float* hn     = ws + off; off += (size_t)BB * DD;
  float* lnout  = h;  // h is dead after the scan

  const size_t dDD = (size_t)DD * DD;
  const size_t dDF = (size_t)DD * FF;
  const size_t dFD = (size_t)FF * DD;

  embed_kernel<<<BB * TT, 256, 0, stream>>>(ids, wte, wpe, h);
  init_states_kernel<<<512, 256, 0, stream>>>(slot_init, states);

  for (int t = 0; t < TT; ++t) {
    gather_x_kernel<<<BB, 256, 0, stream>>>(h, x, t);
    for (int l = 0; l < LL; ++l) {
      float* st = states + (size_t)l * BB * NS * DD;
      // --- write: projections + softmax addressing + EMA state update ---
      launch_gemm(false, false, false, x, Wk + l * dDD, nullptr, kproj, BB, DD, DD, stream);
      launch_gemm(false, false, false, x, Wv + l * dDD, nullptr, vproj, BB, DD, DD, stream);
      launch_gemm(false, false, false, x, Wq + l * dDD, nullptr, q,     BB, DD, DD, stream);
      write_update_kernel<<<BB, 256, 0, stream>>>(st, kproj, vproj);
      // --- think: 3 ERG refinement steps ---
      for (int s = 0; s < 3; ++s) {
        attn_ctx_kernel<<<BB, 256, 0, stream>>>(q, st, st, ctx);
        add_ln_kernel<<<BB, 256, 0, stream>>>(q, ctx, nullptr, q,
                                              ln1_g + l * DD, ln1_b + l * DD);
        launch_gemm(false, true, true,  q,   ffn1_w + l * dDF, ffn1_b + l * FF, ff1, BB, FF, DD, stream);
        launch_gemm(false, true, false, ff1, ffn2_w + l * dFD, ffn2_b + l * DD, ff2, BB, DD, FF, stream);
        add_ln_kernel<<<BB, 256, 0, stream>>>(q, ff2, nullptr, q,
                                              ln2t_g + l * DD, ln2t_b + l * DD);
      }
      // --- read: project slots, attend, residual out_proj ---
      launch_gemm(false, false, false, st, Rk + l * dDD, nullptr, keys, BB * NS, DD, DD, stream);
      launch_gemm(false, false, false, st, Rv + l * dDD, nullptr, vals, BB * NS, DD, DD, stream);
      attn_ctx_kernel<<<BB, 256, 0, stream>>>(q, keys, vals, ctx);
      launch_gemm(false, true, false, ctx, out_w + l * dDD, out_b + l * DD, tmp, BB, DD, DD, stream);
      // x = x + out_proj ; hn = LN(x)
      add_ln_kernel<<<BB, 256, 0, stream>>>(x, tmp, x, hn,
                                            lnm_g + l * DD, lnm_b + l * DD);
      // MLP residual
      launch_gemm(false, true, true,  hn,  mlp1_w + l * dDF, mlp1_b + l * FF, ff1, BB, FF, DD, stream);
      launch_gemm(false, true, false, ff1, mlp2_w + l * dFD, mlp2_b + l * DD, ff2, BB, DD, FF, stream);
      vecadd_kernel<<<(BB * DD + 255) / 256, 256, 0, stream>>>(x, ff2, BB * DD);
    }
    scatter_x_kernel<<<BB, 256, 0, stream>>>(x, outbuf, t);
  }

  // final LN + tied lm_head (W transposed: logits = LN(out) @ wte^T)
  add_ln_kernel<<<BB * TT, 256, 0, stream>>>(outbuf, nullptr, nullptr, lnout, lnf_g, lnf_b);
  launch_gemm(true, false, false, lnout, wte, nullptr, logits, BB * TT, VV, DD, stream);
}
